// Push_info_up_22625887715771
// MI455X (gfx1250) — matrix operand order repaired and checked by reading, not compiled, not run on hardware
//
#include <hip/hip_runtime.h>

typedef __attribute__((ext_vector_type(16))) _Float16 v16h;
typedef __attribute__((ext_vector_type(8)))  _Float16 v8h;
typedef __attribute__((ext_vector_type(8)))  float    v8f;

#define D       256
#define NFAN    5
#define KW1     (D * NFAN)      // 1280
#define N_ABOVE 100000
#define N_DOWN  50000
#define LSTR    264             // f16 hid row stride (halves); 528B, 16B-divisible
#define OSTR    260             // f32 out-staging row stride (floats); 1040B, 16B-divisible
#define BCOLS   40              // B-slice per-column stride (halves); 80B -> 2-pass bank floor
#define BSLICE  (256 * BCOLS)   // halves per B slice buffer (10,240 -> 20,480B)

union ABFrag { v16h v; v8h h[2]; };

__device__ __forceinline__ v8h ld_v8h(const _Float16* p) {
    return *(const v8h*)p;
}

// 4 async b128 copies: INST_OFFSET advances BOTH the LDS and global addresses
// (ISA 08 §4.4), so one base pair covers the 64B column chunk.
__device__ __forceinline__ void async_copy_col64(unsigned ldst, const _Float16* gsrc) {
    asm volatile(
        "global_load_async_to_lds_b128 %0, %1, off\n\t"
        "global_load_async_to_lds_b128 %0, %1, off offset:16\n\t"
        "global_load_async_to_lds_b128 %0, %1, off offset:32\n\t"
        "global_load_async_to_lds_b128 %0, %1, off offset:48"
        :: "v"(ldst), "v"(gsrc) : "memory");
}

// ---------------- prep kernels ----------------

__global__ void zero16_kernel(uint4* __restrict__ p) {
    uint4 z; z.x = 0u; z.y = 0u; z.z = 0u; z.w = 0u;
    p[(size_t)blockIdx.x * 256 + threadIdx.x] = z;
}

__global__ void scatter_h_kernel(const float* __restrict__ h,
                                 const int* __restrict__ idx,
                                 _Float16* __restrict__ newh) {
    int r = blockIdx.x;
    int c = threadIdx.x;
    int drow = idx[r];
    newh[(size_t)drow * D + c] = (_Float16)h[(size_t)r * D + c];
}

// wt[n*K + k] = (f16) w[k*N + n];  N == 256 == blockDim.x
__global__ void transpose_w_kernel(const float* __restrict__ w,
                                   _Float16* __restrict__ wt, int K) {
    int k = blockIdx.x;
    int n = threadIdx.x;
    wt[(size_t)n * K + k] = (_Float16)w[(size_t)k * D + n];
}

// ---------------- fused MLP kernel ----------------
// Block: 256 threads = 8 waves arranged 2 (M) x 4 (N).
// Block tile: 64 rows x 256 cols. Wave tile: 32 rows x 64 cols
//   = 2 M-subtiles (u) x 4 N-subtiles (t) of 16x16 -> 8 accumulators.
// W1^T K-slices (32K x 256N f16) are async-DMA'd into a triple-buffered LDS
// ring: loaded ONCE per block instead of per wave-group.

__global__ __launch_bounds__(256) void fused_mlp_kernel(
    const _Float16* __restrict__ newh,   // [N_ABOVE, 256] f16
    const int*      __restrict__ jidx,   // [N_ABOVE*5]
    const _Float16* __restrict__ w1t,    // [256, 1280] f16 (N-major)
    const float*    __restrict__ b1,     // [256]
    const _Float16* __restrict__ w2t,    // [256, 256] f16 (N-major)
    const float*    __restrict__ b2,     // [256]
    float*          __restrict__ out)    // [N_ABOVE, 256] f32
{
    // 3 B-slice buffers (61,440B) overlaid with hid (33,792B) / f32 staging (33,280B)
    __shared__ alignas(16) _Float16 smh[3 * BSLICE];
    _Float16* hid  = smh;
    float*    smemf = (float*)smh;

    const int tid  = threadIdx.x;
    const int wid  = tid >> 5;
    const int lane = tid & 31;
    const int wm   = wid >> 2;           // 0..1  (M group)
    const int wn   = wid & 3;            // 0..3  (N group)
    const int n0   = wn * 64;
    const int m0   = blockIdx.x * 64 + wm * 32;
    const int c    = lane & 15;          // A: M row within subtile; B/C: N col
    const int g    = lane >> 4;          // K-chunk group select

    const unsigned lds_base = (unsigned)(uintptr_t)&smh[0];

    // ---------------- GEMM1: hid = relu(gather @ W1 + b1) ----------------
    // Gather-row pointers (clamped for the partial last block; clamped rows
    // recompute row N_ABOVE-1 and are never stored).
    const _Float16* aptr[2][NFAN];
    #pragma unroll
    for (int u = 0; u < 2; ++u) {
        int grow = m0 + 16 * u + c;
        if (grow >= N_ABOVE) grow = N_ABOVE - 1;
        #pragma unroll
        for (int f = 0; f < NFAN; ++f)
            aptr[u][f] = newh + (size_t)jidx[grow * NFAN + f] * D + 8 * g;
    }

    // async-copy helper values: thread tid owns column tid of each slice
    const _Float16* bcol = w1t + (size_t)tid * KW1;       // K-contiguous column
    const unsigned  lcol = lds_base + (unsigned)(tid * BCOLS * 2);

    // prologue: issue slices 0 and 1 into ring buffers 0 and 1
    #pragma unroll
    for (int s0 = 0; s0 < 2; ++s0)
        async_copy_col64(lcol + (unsigned)(s0 * BSLICE * 2), bcol + s0 * 32);

    v8f acc[2][4] = {};
    ABFrag a_cur[2];
    #pragma unroll
    for (int u = 0; u < 2; ++u) {
        a_cur[u].h[0] = ld_v8h(aptr[u][0]);
        a_cur[u].h[1] = ld_v8h(aptr[u][0] + 16);
    }

    #pragma unroll
    for (int s = 0; s < 40; ++s) {
        // slice s resident for this wave (4 ops/slice/wave, in-order retire),
        // then block-wide via raw split barrier (no loadcnt/dscnt drain:
        // async writes are ordered by asynccnt; in-flight ds reads always
        // target a different ring buffer than the writes issued next).
        if (s == 39) asm volatile("s_wait_asynccnt 0x0" ::: "memory");
        else         asm volatile("s_wait_asynccnt 0x4" ::: "memory");
        asm volatile("s_barrier_signal -1\n\ts_barrier_wait -1" ::: "memory");

        // issue slice s+2 into ring buffer (s+2)%3
        if (s + 2 < 40)
            async_copy_col64(lcol + (unsigned)(((s + 2) % 3) * BSLICE * 2),
                             bcol + (s + 2) * 32);

        // prefetch next A fragments (register double buffer)
        const int sn = (s + 1) % 40;     // wraps on last iter; 1 wasted prefetch
        const int fn = sn >> 3;
        const int kn = (sn & 7) * 32;
        ABFrag a_nxt[2];
        #pragma unroll
        for (int u = 0; u < 2; ++u) {
            a_nxt[u].h[0] = ld_v8h(aptr[u][fn] + kn);
            a_nxt[u].h[1] = ld_v8h(aptr[u][fn] + kn + 16);
        }

        // B fragments from LDS slice s
        const int bb = (s % 3) * BSLICE;
        ABFrag bf[4];
        #pragma unroll
        for (int t = 0; t < 4; ++t) {
            const _Float16* bp = &smh[bb + (n0 + 16 * t + c) * BCOLS + 8 * g];
            bf[t].h[0] = *(const v8h*)(bp);
            bf[t].h[1] = *(const v8h*)(bp + 16);
        }

        #pragma unroll
        for (int t = 0; t < 4; ++t)
            #pragma unroll
            for (int u = 0; u < 2; ++u)
                acc[u][t] = __builtin_amdgcn_wmma_f32_16x16x32_f16(
                    false, a_cur[u].v, false, bf[t].v, (short)0, acc[u][t], false, false);
        #pragma unroll
        for (int u = 0; u < 2; ++u) a_cur[u] = a_nxt[u];
    }

    __syncthreads();   // all waves done with B ring before hid overlays it

    // bias + relu, stage f16 tile in LDS (rows = block-relative 0..63)
    #pragma unroll
    for (int t = 0; t < 4; ++t) {
        const int col = n0 + 16 * t + c;
        const float bv = b1[col];
        #pragma unroll
        for (int u = 0; u < 2; ++u)
            #pragma unroll
            for (int v = 0; v < 8; ++v) {
                float x = acc[u][t][v] + bv;
                x = x > 0.0f ? x : 0.0f;
                hid[(wm * 32 + 16 * u + 8 * g + v) * LSTR + col] = (_Float16)x;
            }
    }
    __syncthreads();

    // ---------------- GEMM2: out = hid @ W2 + b2 ----------------
    const _Float16* hrow[2];
    #pragma unroll
    for (int u = 0; u < 2; ++u)
        hrow[u] = &hid[(wm * 32 + 16 * u + c) * LSTR + 8 * g];
    const _Float16* b2base[4];
    #pragma unroll
    for (int t = 0; t < 4; ++t)
        b2base[t] = w2t + (size_t)(n0 + 16 * t + c) * D + 8 * g;

    v8f acc2[2][4] = {};
    ABFrag a2_cur[2], c2_cur[4];
    #pragma unroll
    for (int u = 0; u < 2; ++u) {
        a2_cur[u].h[0] = *(const v8h*)(hrow[u]);
        a2_cur[u].h[1] = *(const v8h*)(hrow[u] + 16);
    }
    #pragma unroll
    for (int t = 0; t < 4; ++t) {
        c2_cur[t].h[0] = ld_v8h(b2base[t]);
        c2_cur[t].h[1] = ld_v8h(b2base[t] + 16);
    }

    #pragma unroll
    for (int kk = 0; kk < 8; ++kk) {
        const int kkn = ((kk + 1) & 7) * 32;
        ABFrag a2_nxt[2], c2_nxt[4];
        #pragma unroll
        for (int u = 0; u < 2; ++u) {
            a2_nxt[u].h[0] = *(const v8h*)(hrow[u] + kkn);
            a2_nxt[u].h[1] = *(const v8h*)(hrow[u] + kkn + 16);
        }
        #pragma unroll
        for (int t = 0; t < 4; ++t) {
            c2_nxt[t].h[0] = ld_v8h(b2base[t] + kkn);
            c2_nxt[t].h[1] = ld_v8h(b2base[t] + kkn + 16);
        }
        #pragma unroll
        for (int t = 0; t < 4; ++t)
            #pragma unroll
            for (int u = 0; u < 2; ++u)
                acc2[u][t] = __builtin_amdgcn_wmma_f32_16x16x32_f16(
                    false, a2_cur[u].v, false, c2_cur[t].v, (short)0, acc2[u][t], false, false);
        #pragma unroll
        for (int u = 0; u < 2; ++u) a2_cur[u] = a2_nxt[u];
        #pragma unroll
        for (int t = 0; t < 4; ++t) c2_cur[t] = c2_nxt[t];
    }

    __syncthreads();   // all waves done reading hid before reusing smem as f32

    // ---------------- epilogue: 2 passes of 32 rows, coalesced b128 stores ----
    const int rr   = tid >> 6;           // 0..3
    const int col4 = (tid & 63) * 4;     // 0..252
    #pragma unroll
    for (int pass = 0; pass < 2; ++pass) {
        if (wm == pass) {
            #pragma unroll
            for (int t = 0; t < 4; ++t) {
                const int col = n0 + 16 * t + c;
                const float bv = b2[col];
                #pragma unroll
                for (int u = 0; u < 2; ++u)
                    #pragma unroll
                    for (int v = 0; v < 8; ++v)
                        smemf[(16 * u + 8 * g + v) * OSTR + col] = acc2[u][t][v] + bv;
            }
        }
        __syncthreads();
        #pragma unroll
        for (int p = 0; p < 8; ++p) {
            const int r = rr + p * 4;
            const int grow = blockIdx.x * 64 + pass * 32 + r;
            if (grow < N_ABOVE) {
                float4 vv = *(const float4*)&smemf[r * OSTR + col4];
                *(float4*)&out[(size_t)grow * D + col4] = vv;
            }
        }
        __syncthreads();
    }
}

// ---------------- launcher ----------------

extern "C" void kernel_launch(void* const* d_in, const int* in_sizes, int n_in,
                              void* d_out, int out_size, void* d_ws, size_t ws_size,
                              hipStream_t stream) {
    const float* h   = (const float*)d_in[0];
    // d_in[1] = h_above (only shape used by reference)
    const int*   idx = (const int*)d_in[2];
    // d_in[3] = i (structure only)
    const int*   jj  = (const int*)d_in[4];
    const float* W1  = (const float*)d_in[5];
    const float* b1  = (const float*)d_in[6];
    const float* W2  = (const float*)d_in[7];
    const float* b2  = (const float*)d_in[8];
    float* out = (float*)d_out;

    char* ws = (char*)d_ws;
    _Float16* newh = (_Float16*)ws;                             // 100000*256*2 = 51,200,000 B
    _Float16* w1t  = (_Float16*)(ws + 51200000);                // 1280*256*2   =    655,360 B
    _Float16* w2t  = (_Float16*)(ws + 51200000 + 655360);       // 256*256*2    =    131,072 B

    // new_h = scatter(h, idx) in f16
    zero16_kernel<<<12500, 256, 0, stream>>>((uint4*)newh);     // 12500*256*16B = 51.2 MB exact
    scatter_h_kernel<<<N_DOWN, 256, 0, stream>>>(h, idx, newh);
    // f16 transposed weights (K-contiguous per output column -> B-fragment friendly)
    transpose_w_kernel<<<KW1, 256, 0, stream>>>(W1, w1t, KW1);
    transpose_w_kernel<<<D,   256, 0, stream>>>(W2, w2t, D);

    // fused gather-GEMM1-relu-GEMM2; 64-row block tiles, clamp-guarded tail
    const int nblk = (N_ABOVE + 63) / 64;   // 1563
    fused_mlp_kernel<<<nblk, 256, 0, stream>>>(newh, jj, w1t, b1, w2t, b2, out);
}
